// WindowAttention_26955214749844
// MI455X (gfx1250) — matrix operand order repaired and checked by reading, not compiled, not run on hardware
//
#include <hip/hip_runtime.h>

typedef _Float16 v16h __attribute__((ext_vector_type(16)));
typedef _Float16 v8h  __attribute__((ext_vector_type(8)));
typedef float    v8f  __attribute__((ext_vector_type(8)));

#define NTOK 64
#define CDIM 384
#define TC   1152
#define NH   12
#define DH   32
#define NWAVE 16
#define LAMBDA_INIT 0.3555090676f
#define QSCALE 0.17677669529663687f   // 32^-0.5

// ---------- WMMA fragment helpers (CDNA5 wave32 layouts, ISA 7.12.2) ----------

// A-matrix 16x32 (MxK), 16-bit: lanes 0-15 hold M=0..15 with K {0..7,16..23},
// lanes 16-31 hold same rows with K {8..15,24..31}.
__device__ __forceinline__ v16h frag_a(const _Float16* __restrict__ base,
                                       int rstride, int kstride) {
  const int lane = threadIdx.x & 31;
  const int r = lane & 15, hi = lane >> 4;
  const _Float16* p = base + r * rstride;
  v16h a;
#pragma unroll
  for (int e = 0; e < 8; ++e) a[e] = p[(hi * 8 + e) * kstride];
#pragma unroll
  for (int e = 0; e < 8; ++e) a[8 + e] = p[(16 + hi * 8 + e) * kstride];
  return a;
}

// B-matrix 32x16 (KxN), 16-bit: lanes 0-15 hold col N=lane with K {0..7,16..23},
// lanes 16-31 hold col N=lane-16 with K {8..15,24..31} (same K order as A).
__device__ __forceinline__ v16h frag_b(const _Float16* __restrict__ base,
                                       int kstride, int cstride) {
  const int lane = threadIdx.x & 31;
  const int c = lane & 15, hi = lane >> 4;
  const _Float16* p = base + c * cstride;
  v16h b;
#pragma unroll
  for (int e = 0; e < 8; ++e) b[e] = p[(hi * 8 + e) * kstride];
#pragma unroll
  for (int e = 0; e < 8; ++e) b[8 + e] = p[(16 + hi * 8 + e) * kstride];
  return b;
}

__device__ __forceinline__ v8f wmma_f16(v16h a, v16h b, v8f c) {
  return __builtin_amdgcn_wmma_f32_16x16x32_f16(false, a, false, b,
                                                (short)0, c, false, false);
}

// branchless ELU+1 feature map: f(x) = x+1 (x>=0) else exp(x)
__device__ __forceinline__ float fmap(float v) {
  float ev = __expf(v);                 // computed unconditionally -> v_cndmask
  return (v >= 0.f) ? (v + 1.f) : ev;
}

// ---------- prep: transpose weights to f16, per-head lambda scalar ----------

__global__ void prep_kernel(const float* __restrict__ qkv_w,
                            const float* __restrict__ proj_w,
                            const float* __restrict__ lk1, const float* __restrict__ lv1,
                            const float* __restrict__ lk2, const float* __restrict__ lv2,
                            _Float16* __restrict__ WqkvT, _Float16* __restrict__ projT,
                            float* __restrict__ lamS) {
  const int idx = blockIdx.x * blockDim.x + threadIdx.x;
  const int T1 = TC * CDIM;
  const int T2 = CDIM * CDIM;
  if (idx < T1) {
    int oc = idx / CDIM, ic = idx - oc * CDIM;
    WqkvT[idx] = (_Float16)qkv_w[ic * TC + oc];          // [out_col][in_k]
  } else if (idx < T1 + T2) {
    int j = idx - T1;
    int oc = j / CDIM, ic = j - oc * CDIM;
    projT[j] = (_Float16)proj_w[ic * CDIM + oc];
  }
  if (idx < NH) {
    float s1 = 0.f, s2 = 0.f;
    for (int dd = 0; dd < DH; ++dd) {
      s1 += lk1[idx * DH + dd] * lv1[idx * DH + dd];
      s2 += lk2[idx * DH + dd] * lv2[idx * DH + dd];
    }
    float lam_full = __expf(s1) - __expf(s2) + LAMBDA_INIT;
    lamS[idx] = 1.0f - lam_full;                          // attn = (1-lambda)*A
  }
}

// ---------- fused per-batch kernel ----------
// LDS (224KB, one WG/WGP):
//   qs  64x384  f16 row-major       (48KB)   q, scaled
//   kT  384x64  f16  kT[c][n]       (48KB)   feature-mapped k, transposed
//   vT  384x64  f16  vT[c][n]       (48KB)   feature-mapped v, transposed
//   ao  64x384  f16 row-major       (48KB)   x tile, then attn output
//   atT 16 waves * 32x32 f16        (32KB)   per-head attn matrix, transposed

__global__ void __launch_bounds__(512, 1)
fused_attn_kernel(const float* __restrict__ x, const float* __restrict__ qkv_b,
                  const float* __restrict__ proj_b,
                  const _Float16* __restrict__ WqkvT, const _Float16* __restrict__ projT,
                  const float* __restrict__ lamS, float* __restrict__ out) {
  extern __shared__ char smem_raw[];
  _Float16* qs   = (_Float16*)smem_raw;            // 64*384
  _Float16* kT   = qs  + NTOK * CDIM;              // 384*64
  _Float16* vT   = kT  + CDIM * NTOK;              // 384*64
  _Float16* ao   = vT  + CDIM * NTOK;              // 64*384 (x f16 -> attn out)
  _Float16* abuf = ao  + NTOK * CDIM;              // 16 * 32*32

  const int b    = blockIdx.x;
  const int tid  = threadIdx.x;
  const int wave = tid >> 5;
  const int lane = tid & 31;
  const int lc   = lane & 15, lhi = lane >> 4;

  // ---- stage 0: x tile -> f16 LDS (ao region) ----
  const float4* xv = (const float4*)(x + (size_t)b * NTOK * CDIM);
  for (int i = tid; i < NTOK * CDIM / 4; i += 512) {
    float4 t = xv[i];
    ao[4 * i + 0] = (_Float16)t.x; ao[4 * i + 1] = (_Float16)t.y;
    ao[4 * i + 2] = (_Float16)t.z; ao[4 * i + 3] = (_Float16)t.w;
  }
  __syncthreads();

  // ---- stage 1: qkv = x @ Wqkv + b; fused scale / feature-map / transpose ----
  for (int tile = wave; tile < 4 * (TC / 16); tile += NWAVE) {
    const int rt = tile / (TC / 16);
    const int ct = tile - rt * (TC / 16);
    v8f acc = {};
#pragma unroll
    for (int kk = 0; kk < CDIM; kk += 32) {
      v16h a  = frag_a(ao + rt * 16 * CDIM + kk, CDIM, 1);
      v16h bb = frag_b(WqkvT + ct * 16 * CDIM + kk, 1, CDIM);
      acc = wmma_f16(a, bb, acc);
    }
    const int col  = ct * 16 + lc;               // global qkv column
    const float bias = qkv_b[col];
    const int rbase = rt * 16 + lhi * 8;         // 8 consecutive rows per lane
    if (ct < CDIM / 16) {                        // ---- q: scale, row-major, strided
#pragma unroll
      for (int r = 0; r < 8; ++r)
        qs[(rbase + r) * CDIM + col] = (_Float16)((acc[r] + bias) * QSCALE);
    } else if (ct < 2 * (CDIM / 16)) {           // ---- k: fmap, transposed b128 store
      v8h pk;
#pragma unroll
      for (int r = 0; r < 8; ++r) pk[r] = (_Float16)fmap(acc[r] + bias);
      *(v8h*)(kT + (col - CDIM) * NTOK + rbase) = pk;
    } else {                                     // ---- v: fmap, transposed b128 store
      v8h pk;
#pragma unroll
      for (int r = 0; r < 8; ++r) pk[r] = (_Float16)fmap(acc[r] + bias);
      *(v8h*)(vT + (col - 2 * CDIM) * NTOK + rbase) = pk;
    }
  }
  __syncthreads();

  // ---- stage 2: per head, A = (fk @ fv) .* gauss .* (1-lambda); out = q @ A ----
  for (int h = wave; h < NH; h += NWAVE) {
    _Float16* atT = abuf + wave * (DH * DH);     // attn^T: atT[e][d]
    const float gs = lamS[h];
#pragma unroll
    for (int mt = 0; mt < 2; ++mt)
#pragma unroll
      for (int nt = 0; nt < 2; ++nt) {
        v8f acc = {};
#pragma unroll
        for (int kk = 0; kk < NTOK; kk += 32) {
          // fk rows d (contiguous n): kT[(h*32+d)*64 + n]
          v16h a  = frag_a(kT + (h * DH + mt * 16) * NTOK + kk, NTOK, 1);
          // fv cols d (contiguous n): vT[(h*32+d)*64 + n]
          v16h bb = frag_b(vT + (h * DH + nt * 16) * NTOK + kk, 1, NTOK);
          acc = wmma_f16(a, bb, acc);
        }
        const int ccol = nt * 16 + lc;           // e index
        v8h pk;
#pragma unroll
        for (int r = 0; r < 8; ++r) {
          const int crow = mt * 16 + r + lhi * 8; // d index
          const float diff = (float)(crow - ccol);
          pk[r] = (_Float16)(acc[r] * __expf(-2.0f * diff * diff) * gs); // sigma=0.5
        }
        // transposed store: atT[e*32 + d], 8 consecutive d -> one b128
        *(v8h*)(atT + ccol * DH + mt * 16 + lhi * 8) = pk;
      }
    // out_h = q(64x32) @ attn(32x32) -> ao[:, h*32 : h*32+32]
#pragma unroll
    for (int mt = 0; mt < 4; ++mt)
#pragma unroll
      for (int nt = 0; nt < 2; ++nt) {
        v16h a  = frag_a(qs + mt * 16 * CDIM + h * DH, CDIM, 1); // contiguous K=d
        v16h bb = frag_b(atT + nt * 16 * DH, 1, DH);             // contiguous K=d
        v8f acc = {};
        acc = wmma_f16(a, bb, acc);
        const int ccol = nt * 16 + lc;
#pragma unroll
        for (int r = 0; r < 8; ++r) {
          const int crow = mt * 16 + r + lhi * 8;
          ao[crow * CDIM + h * DH + ccol] = (_Float16)acc[r];
        }
      }
  }
  __syncthreads();

  // ---- stage 3: out = attn_out @ proj_w + proj_b (f32 store) ----
  float* outb = out + (size_t)b * NTOK * CDIM;
  for (int tile = wave; tile < 4 * (CDIM / 16); tile += NWAVE) {
    const int rt = tile / (CDIM / 16);
    const int ct = tile - rt * (CDIM / 16);
    v8f acc = {};
#pragma unroll
    for (int kk = 0; kk < CDIM; kk += 32) {
      v16h a  = frag_a(ao + rt * 16 * CDIM + kk, CDIM, 1);
      v16h bb = frag_b(projT + ct * 16 * CDIM + kk, 1, CDIM);
      acc = wmma_f16(a, bb, acc);
    }
    const int col = ct * 16 + lc;
    const float bias = proj_b[col];
#pragma unroll
    for (int r = 0; r < 8; ++r) {
      const int row = rt * 16 + r + lhi * 8;
      outb[row * CDIM + col] = acc[r] + bias;
    }
  }
}

// ---------- host launch ----------

extern "C" void kernel_launch(void* const* d_in, const int* in_sizes, int n_in,
                              void* d_out, int out_size, void* d_ws, size_t ws_size,
                              hipStream_t stream) {
  (void)n_in; (void)out_size; (void)ws_size;
  const float* x      = (const float*)d_in[0];
  const float* qkv_w  = (const float*)d_in[1];
  const float* qkv_b  = (const float*)d_in[2];
  const float* proj_w = (const float*)d_in[3];
  const float* proj_b = (const float*)d_in[4];
  const float* lk1    = (const float*)d_in[5];
  const float* lv1    = (const float*)d_in[6];
  const float* lk2    = (const float*)d_in[7];
  const float* lv2    = (const float*)d_in[8];
  float* out = (float*)d_out;

  // workspace layout: WqkvT f16 [1152*384] | projT f16 [384*384] | lamS f32[12]
  _Float16* WqkvT = (_Float16*)d_ws;
  _Float16* projT = WqkvT + TC * CDIM;
  float*    lamS  = (float*)(projT + CDIM * CDIM);

  const int prep_elems = TC * CDIM + CDIM * CDIM;
  prep_kernel<<<(prep_elems + 255) / 256, 256, 0, stream>>>(
      qkv_w, proj_w, lk1, lv1, lk2, lv2, WqkvT, projT, lamS);

  const int B = in_sizes[0] / (NTOK * CDIM);   // 2048
  const size_t shmem =
      (size_t)(NTOK * CDIM        // qs
             + CDIM * NTOK        // kT
             + CDIM * NTOK        // vT
             + NTOK * CDIM        // ao
             + NWAVE * DH * DH)   // atT scratch
      * sizeof(_Float16);         // 224KB
  hipFuncSetAttribute((const void*)fused_attn_kernel,
                      hipFuncAttributeMaxDynamicSharedMemorySize, (int)shmem);
  fused_attn_kernel<<<B, 512, shmem, stream>>>(x, qkv_b, proj_b,
                                               WqkvT, projT, lamS, out);
}
